// HashGrid_w_pose_28544352649908
// MI455X (gfx1250) — matrix operand order repaired and checked by reading, not compiled, not run on hardware
//
#include <hip/hip_runtime.h>
#include <stdint.h>

typedef __attribute__((ext_vector_type(16))) _Float16 v16h;
typedef __attribute__((ext_vector_type(8)))  _Float16 v8h;
typedef __attribute__((ext_vector_type(2)))  _Float16 h2v;
typedef __attribute__((ext_vector_type(8)))  float    v8f;
typedef __attribute__((ext_vector_type(2)))  float    v2f;

#define LVL   16
#define TSIZE (1u << 19)
#define HID   64
#define NOUT  9
#define BLK   128      // threads per block (4 waves)
#define PTS   128      // points per block

// LDS row strides in f16 elements (byte strides multiples of 16, conflict-free)
#define SENC 40        // enc rows: 32 used
#define SHID 72        // hidden rows: 64 used
#define SW1  40        // W1^T rows: k-dim 32
#define SW2  72        // W2^T rows: k-dim 64
#define SW3  72        // W3^T rows: k-dim 64

__device__ __forceinline__ v8f wmma16(v16h a, v16h b, v8f c) {
  return __builtin_amdgcn_wmma_f32_16x16x32_f16(false, a, false, b, (short)0, c, false, false);
}

// packed f32x2 -> f16x2 convert (v_cvt_pk_rtz_f16_f32), re-typed to h2v
__device__ __forceinline__ h2v cvt_pk(float a, float b) {
  return __builtin_bit_cast(h2v, __builtin_amdgcn_cvt_pkrtz(a, b));
}

// A-fragment (16x32 f16) from row-major LDS buffer.
// lane r=lane&15 is row M, kh=lane>>4; element e -> K=(e>>3)*16 + kh*8 + (e&7)
__device__ __forceinline__ v16h frag_a(const _Float16* buf, int strideH, int row0, int k0, int lane) {
  const int r  = lane & 15;
  const int kh = lane >> 4;
  const _Float16* p = buf + (row0 + r) * strideH + k0 + kh * 8;
  v8h lo = *(const v8h*)(p);        // K = k0+kh*8    .. +7
  v8h hi = *(const v8h*)(p + 16);   // K = k0+16+kh*8 .. +7
  v16h a;
#pragma unroll
  for (int i = 0; i < 8; ++i) { a[i] = lo[i]; a[i + 8] = hi[i]; }
  return a;
}

// B-fragment (32x16 f16) from transposed weights Wt[n][k].
// lane holds column N=lane&15; element e -> K = k0 + (e>>3)*16 + (lane>>4)*8 + (e&7)
__device__ __forceinline__ v16h frag_b(const _Float16* wt, int strideK, int n0, int k0, int lane) {
  const int n  = lane & 15;
  const int kh = lane >> 4;
  const _Float16* p = wt + (n0 + n) * strideK + k0 + kh * 8;
  v8h lo = *(const v8h*)(p);
  v8h hi = *(const v8h*)(p + 16);
  v16h b;
#pragma unroll
  for (int i = 0; i < 8; ++i) { b[i] = lo[i]; b[i + 8] = hi[i]; }
  return b;
}

// f16 convert (packed) + packed-f16 ReLU + store 8 C/D elements of one tile column.
// cvt result is canonical, so maxnum lowers to a single v_pk_max_num_f16.
__device__ __forceinline__ void relu_store8(_Float16* dst0, int strideH, v8f acc) {
  const h2v z = {(_Float16)0.0f, (_Float16)0.0f};
#pragma unroll
  for (int j = 0; j < 8; j += 2) {
    h2v p = cvt_pk(acc[j], acc[j + 1]);       // v_cvt_pk_rtz_f16_f32
    p = __builtin_elementwise_max(p, z);      // v_pk_max_num_f16
    dst0[j * strideH]       = p.x;            // ds_store_b16
    dst0[(j + 1) * strideH] = p.y;            // ds_store_b16_d16_hi
  }
}

__global__ void __launch_bounds__(BLK)
hashgrid_mlp_fused(const float* __restrict__ img,   // [N,2]
                   const float* __restrict__ pose,  // [N,2]
                   const float* __restrict__ table, // [L*T,2]
                   const float* __restrict__ W1,    // [32,64]
                   const float* __restrict__ W2,    // [64,64]
                   const float* __restrict__ W3,    // [64,9]
                   float* __restrict__ out,         // [N,9]
                   int npts) {
  __shared__ _Float16 sEnc[PTS * SENC];
  __shared__ _Float16 sHid[PTS * SHID];
  __shared__ _Float16 sW1t[64 * SW1];
  __shared__ _Float16 sW2t[64 * SW2];
  __shared__ _Float16 sW3t[16 * SW3];

  const int tid = threadIdx.x;

  // ---- cooperative weight load (transposed, f32 -> f16) ----
  for (int i = tid; i < 32 * 64; i += BLK) {
    int k = i >> 6, n = i & 63;
    sW1t[n * SW1 + k] = (_Float16)W1[i];
  }
  for (int i = tid; i < 64 * 64; i += BLK) {
    int k = i >> 6, n = i & 63;
    sW2t[n * SW2 + k] = (_Float16)W2[i];
  }
  for (int i = tid; i < 16 * 64; i += BLK) {
    int n = i >> 6, k = i & 63;
    sW3t[n * SW3 + k] = (n < NOUT) ? (_Float16)W3[k * NOUT + n] : (_Float16)0.0f;
  }

  // ---- phase 1: hash-grid encode, one point per thread ----
  {
    const int gp   = blockIdx.x * PTS + tid;
    const int pidx = (gp < npts) ? gp : (npts - 1);   // clamp, keep EXEC full
    const float x0 = img[2 * pidx],  x1 = img[2 * pidx + 1];
    const float x2 = pose[2 * pidx], x3 = pose[2 * pidx + 1];
    const uint32_t P1 = 2654435761u, P2 = 805459861u, P3 = 3674653429u;
    // all hash components pre-shifted by 3 (float2 = 8 bytes per entry);
    // shifts/AND distribute over XOR, so index math stays exact mod 2^32
    const uint32_t M8 = (TSIZE - 1u) << 3;
    const char* __restrict__ tb = (const char*)table;

    float res = 16.0f;
#pragma unroll
    for (int l = 0; l < LVL; ++l) {
      const float s = floorf(res);
      float q, g;
      q = x0 * s; g = floorf(q); const float f0 = q - g; const uint32_t u0 = (uint32_t)g;
      q = x1 * s; g = floorf(q); const float f1 = q - g; const uint32_t u1 = (uint32_t)g;
      q = x2 * s; g = floorf(q); const float f2 = q - g; const uint32_t u2 = (uint32_t)g;
      q = x3 * s; g = floorf(q); const float f3 = q - g; const uint32_t u3 = (uint32_t)g;

      const uint32_t e0 = u0 << 3,          e0b = e0 + 8u;
      const uint32_t e1 = (u1 * P1) << 3,   e1b = e1 + (P1 << 3);
      const uint32_t e2 = (u2 * P2) << 3,   e2b = e2 + (P2 << 3);
      const uint32_t e3 = (u3 * P3) << 3,   e3b = e3 + (P3 << 3);

      // pre-masked XOR halves: per corner a single XOR yields the byte offset
      const uint32_t h01[4] = { (e0 ^ e1) & M8,  (e0b ^ e1) & M8,
                                (e0 ^ e1b) & M8, (e0b ^ e1b) & M8 };
      const uint32_t h23[4] = { (e2 ^ e3)  & M8, (e2b ^ e3)  & M8,
                                (e2 ^ e3b) & M8, (e2b ^ e3b) & M8 };
      // level offset is uniform -> fold into the scalar gather base (SADDR)
      const char* __restrict__ tbl = tb + (size_t)((uint32_t)l * (TSIZE * 8u));

      const float w0a = 1.0f - f0, w1a = 1.0f - f1, w2a = 1.0f - f2, w3a = 1.0f - f3;
      // packed weight tables: v_pk_mul_f32
      const v2f b01 = {w0a, f0};
      const v2f w01lo = b01 * w1a;            // {w01[0], w01[1]}
      const v2f w01hi = b01 * f1;             // {w01[2], w01[3]}
      const v2f b23 = {w2a, f2};
      const v2f w23lo = b23 * w3a;            // {w23[0], w23[1]}
      const v2f w23hi = b23 * f3;             // {w23[2], w23[3]}
      const float w23s[4] = { w23lo.x, w23lo.y, w23hi.x, w23hi.y };

      v2f acc = {0.0f, 0.0f};
#pragma unroll
      for (int jj = 0; jj < 4; ++jj) {
        // 4 corner scales for this jj via two pk muls
        const v2f wa = w01lo * w23s[jj];
        const v2f wb = w01hi * w23s[jj];
        const float ws[4] = { wa.x, wa.y, wb.x, wb.y };
        const uint32_t hj = h23[jj];
#pragma unroll
        for (int ii = 0; ii < 4; ++ii) {
          const uint32_t offs = h01[ii] ^ hj;          // single XOR -> byte offset
          v2f f = *(const v2f*)(tbl + offs);           // global_load_b64 (SADDR form)
          acc += f * ws[ii];                           // v_pk_fma_f32
        }
      }
      // packed convert + single b32 store (2*l, 2*l+1 contiguous, 4B aligned)
      h2v ep = cvt_pk(acc.x, acc.y);
      *(h2v*)&sEnc[tid * SENC + 2 * l] = ep;
      res *= 1.38f;
    }
  }
  __syncthreads();

  // ---- phase 2: WMMA MLP, each wave does two 16-row tiles of its own rows ----
  const int wave = tid >> 5;
  const int lane = tid & 31;
  const int nCol = lane & 15;
  const int kh   = lane >> 4;
  const bool blockFull = (blockIdx.x * PTS + PTS) <= npts;   // uniform

  for (int t = 0; t < 2; ++t) {
    const int row0 = wave * 32 + t * 16;

    // Layer 1: enc[16,32] @ W1[32,64]  (K=32 -> single wmma per N-tile)
    v16h aE = frag_a(sEnc, SENC, row0, 0, lane);
    v8f h1[4];
#pragma unroll
    for (int nt = 0; nt < 4; ++nt) {
      v16h b = frag_b(sW1t, SW1, nt * 16, 0, lane);
      v8f z = {};
      h1[nt] = wmma16(aE, b, z);
    }
    // ReLU -> sHid (f16). C/D layout: VGPR j -> M=j+8*kh, N=nCol
#pragma unroll
    for (int nt = 0; nt < 4; ++nt)
      relu_store8(&sHid[(row0 + 8 * kh) * SHID + nt * 16 + nCol], SHID, h1[nt]);

    // Layer 2: hid[16,64] @ W2[64,64]  (K=64 -> 2 accumulating wmmas per N-tile)
    v16h aH0 = frag_a(sHid, SHID, row0, 0, lane);
    v16h aH1 = frag_a(sHid, SHID, row0, 32, lane);
    v8f h2[4];
#pragma unroll
    for (int nt = 0; nt < 4; ++nt) {
      v16h b0 = frag_b(sW2t, SW2, nt * 16, 0, lane);
      v16h b1 = frag_b(sW2t, SW2, nt * 16, 32, lane);
      v8f z = {};
      v8f c = wmma16(aH0, b0, z);
      h2[nt] = wmma16(aH1, b1, c);
    }
    // ReLU -> sHid (overwrite same rows; same-wave LDS ordering)
#pragma unroll
    for (int nt = 0; nt < 4; ++nt)
      relu_store8(&sHid[(row0 + 8 * kh) * SHID + nt * 16 + nCol], SHID, h2[nt]);

    // Layer 3: hid[16,64] @ W3pad[64,16]  (only first 9 cols valid)
    v16h aO0 = frag_a(sHid, SHID, row0, 0, lane);
    v16h aO1 = frag_a(sHid, SHID, row0, 32, lane);
    v16h b0 = frag_b(sW3t, SW3, 0, 0, lane);
    v16h b1 = frag_b(sW3t, SW3, 0, 32, lane);
    v8f z = {};
    v8f o = wmma16(aO0, b0, z);
    o = wmma16(aO1, b1, o);

    // store: row M = j+8*kh, col N = nCol (< 9)
    if (nCol < NOUT) {
      const int base = blockIdx.x * PTS + row0 + 8 * kh;
      if (blockFull) {
#pragma unroll
        for (int j = 0; j < 8; ++j)
          out[(base + j) * NOUT + nCol] = o[j];
      } else {
#pragma unroll
        for (int j = 0; j < 8; ++j) {
          const int r = base + j;
          if (r < npts) out[r * NOUT + nCol] = o[j];
        }
      }
    }
  }
}

extern "C" void kernel_launch(void* const* d_in, const int* in_sizes, int n_in,
                              void* d_out, int out_size, void* d_ws, size_t ws_size,
                              hipStream_t stream) {
  const float* img   = (const float*)d_in[0];
  const float* pose  = (const float*)d_in[1];
  const float* table = (const float*)d_in[2];
  const float* W1    = (const float*)d_in[3];
  const float* W2    = (const float*)d_in[4];
  const float* W3    = (const float*)d_in[5];
  float* out = (float*)d_out;

  const int npts = in_sizes[0] / 2;            // img_pixel_indices is [N,2]
  const int blocks = (npts + PTS - 1) / PTS;   // 1024 for N=131072

  hashgrid_mlp_fused<<<blocks, BLK, 0, stream>>>(img, pose, table, W1, W2, W3, out, npts);
}
